// Sequence_32246614459056
// MI455X (gfx1250) — matrix operand order repaired and checked by reading, not compiled, not run on hardware
//
#include <hip/hip_runtime.h>
#include <hip/hip_bf16.h>

// ---------------------------------------------------------------------------
// 2-layer LSTM (H=100) + linear head, B=4096, T=1024, IN=7.
// 256 WGs x 160 threads (5 waves). Each WG owns 16 batch rows for all 1024
// timesteps; fp16 weights resident in LDS; gates via v_wmma_f32_16x16x32_f16.
// 5 waves x 5 N-tiles = 25 tiles of the 16x400 gate matrix. Biases are
// injected through the WMMA C-matrix; B-fragments are preloaded per tile so
// DS waits graduate instead of draining to zero before every WMMA.
// ---------------------------------------------------------------------------

#define B_   4096
#define T_   1024
#define IN_  7
#define H_   100
#define NG_  400          // 4*H gates
#define K1_  128          // pad(7+100) to 32-mult
#define K2_  224          // pad(100+100) to 32-mult
#define ROWS 16           // batch rows per workgroup
#define NT_  25           // 400/16 gate column tiles
#define WGT  160          // threads per workgroup (5 waves)
#define TPW  5            // N-tiles per wave

typedef __attribute__((ext_vector_type(16))) _Float16 v16h;
typedef __attribute__((ext_vector_type(8)))  _Float16 v8h;
typedef __attribute__((ext_vector_type(8)))  float    v8f;

// ---- LDS layout (bytes) ----------------------------------------------------
#define OFF_W1   0                         // 400*128 fp16 = 102400
#define OFF_W2   102400                    // 400*224 fp16 = 179200
#define OFF_A1   281600                    // 16*128 fp16  = 4096
#define OFF_A2   285696                    // 16*224 fp16  = 7168
#define OFF_G    292864                    // 16*400 f32   = 25600
#define OFF_B1   318464                    // 400 f32      = 1600
#define OFF_B2   320064                    // 400 f32      = 1600
#define OFF_WL   321664                    // 112 f32      = 448 ([100]=b_lin)
#define OFF_RED  322112                    // 160 f32      = 640
#define SMEM_BYTES 322752                  // < 327680 (320KB per WG)

#define LOG2E 1.44269504088896f

// Branchless fast sigmoid/tanh on v_exp_f32 / v_rcp_f32.
__device__ __forceinline__ float fast_sigmoid(float v) {
    float e = __builtin_amdgcn_exp2f(v * -LOG2E);
    return __builtin_amdgcn_rcpf(1.0f + e);
}
__device__ __forceinline__ float fast_tanh(float v) {
    v = fminf(15.0f, fmaxf(-15.0f, v));              // avoid inf/inf
    float e = __builtin_amdgcn_exp2f(v * (2.0f * LOG2E));
    return (e - 1.0f) * __builtin_amdgcn_rcpf(e + 1.0f);
}

// One 16xK @ KxNG tile-GEMM; wave w computes N-tiles [w*5, w*5+5).
// A: row-major [16][STRIDE] fp16 LDS.  W: row-major [400][STRIDE] fp16 LDS
// (B fragment column n == row n of W, i.e. gates = A @ W^T + bias).
// Gate bias (constant per N column) rides in as the C-matrix initial value.
template <int STRIDE, int KC>
__device__ __forceinline__ void gemm_gates(const _Float16* __restrict__ A,
                                           const _Float16* __restrict__ W,
                                           const float* __restrict__ BIAS,
                                           float* __restrict__ G,
                                           int lane, int wave) {
    const int m  = lane & 15;
    const int hs = lane >> 4;            // lane half select
    // A fragments: loaded once, reused across all 5 N-tiles.
    v16h afrag[KC];
#pragma unroll
    for (int kc = 0; kc < KC; ++kc) {
        // half 0 -> K {0..7,16..23}, half 1 -> K {8..15,24..31}
        const _Float16* Ab = A + m * STRIDE + kc * 32 + hs * 8;
        v8h lo = *(const v8h*)(Ab);
        v8h hi = *(const v8h*)(Ab + 16);
        afrag[kc] = __builtin_shufflevector(lo, hi,
            0,1,2,3,4,5,6,7,8,9,10,11,12,13,14,15);
    }
#pragma unroll
    for (int i = 0; i < TPW; ++i) {
        const int nt   = wave * TPW + i;
        const int ncol = nt * 16 + m;
        // bias for this gate column (same for all 8 M-rows held by the lane)
        float bv = BIAS[ncol];
        // preload ALL B fragments of this tile -> graduated s_wait_dscnt
        const _Float16* Wb = W + ncol * STRIDE + hs * 8;
        v16h bfrag[KC];
#pragma unroll
        for (int kc = 0; kc < KC; ++kc) {
            v8h blo = *(const v8h*)(Wb + kc * 32);
            v8h bhi = *(const v8h*)(Wb + kc * 32 + 16);
            bfrag[kc] = __builtin_shufflevector(blo, bhi,
                0,1,2,3,4,5,6,7,8,9,10,11,12,13,14,15);
        }
        v8f acc;
#pragma unroll
        for (int r = 0; r < 8; ++r) acc[r] = bv;
#pragma unroll
        for (int kc = 0; kc < KC; ++kc)
            acc = __builtin_amdgcn_wmma_f32_16x16x32_f16(
                false, afrag[kc], false, bfrag[kc], (short)0, acc, false, false);
        // D layout: VGPR r -> M = r + 8*laneHalf, N = ncol
        const int mb = hs * 8;
#pragma unroll
        for (int r = 0; r < 8; ++r)
            G[(mb + r) * NG_ + ncol] = acc[r];
    }
}

// ---------------------------------------------------------------------------
// Weight packing: fp16, K-padded, input+recurrent weights concatenated.
// W1c[n][k] : k<7 -> W_ih1[n][k], k<107 -> W_hh1[n][k-7], else 0     (K1_=128)
// W2c[n][k] : k<100 -> W_ih2[n][k], k<200 -> W_hh2[n][k-100], else 0 (K2_=224)
// ---------------------------------------------------------------------------
__global__ void prep_kernel(const float* __restrict__ Wih1, const float* __restrict__ Whh1,
                            const float* __restrict__ bih1, const float* __restrict__ bhh1,
                            const float* __restrict__ Wih2, const float* __restrict__ Whh2,
                            const float* __restrict__ bih2, const float* __restrict__ bhh2,
                            const float* __restrict__ Wlin, const float* __restrict__ blin,
                            _Float16* __restrict__ W1c, _Float16* __restrict__ W2c,
                            float* __restrict__ b1, float* __restrict__ b2,
                            float* __restrict__ wl) {
    int idx = blockIdx.x * 256 + threadIdx.x;
    if (idx < NG_ * K1_) {
        int n = idx >> 7, k = idx & 127;
        float v = 0.0f;
        if (k < IN_)            v = Wih1[n * IN_ + k];
        else if (k < IN_ + H_)  v = Whh1[n * H_ + (k - IN_)];
        W1c[idx] = (_Float16)v;
        return;
    }
    idx -= NG_ * K1_;
    if (idx < NG_ * K2_) {
        int n = idx / K2_, k = idx - n * K2_;
        float v = 0.0f;
        if (k < H_)             v = Wih2[n * H_ + k];
        else if (k < 2 * H_)    v = Whh2[n * H_ + (k - H_)];
        W2c[idx] = (_Float16)v;
        return;
    }
    idx -= NG_ * K2_;
    if (idx < NG_) { b1[idx] = bih1[idx] + bhh1[idx]; return; }
    idx -= NG_;
    if (idx < NG_) { b2[idx] = bih2[idx] + bhh2[idx]; return; }
    idx -= NG_;
    if (idx < H_)  { wl[idx] = Wlin[idx]; return; }
    idx -= H_;
    if (idx == 0)  { wl[H_] = blin[0]; }
}

// ---------------------------------------------------------------------------
// Main persistent LSTM kernel.
// ---------------------------------------------------------------------------
__global__ __launch_bounds__(WGT, 1)
void lstm_kernel(const float* __restrict__ x,
                 const _Float16* __restrict__ Wpack,   // W1c || W2c contiguous
                 const float* __restrict__ b1g, const float* __restrict__ b2g,
                 const float* __restrict__ wlg,
                 float* __restrict__ out) {
    extern __shared__ char smem[];
    _Float16* sW1 = (_Float16*)(smem + OFF_W1);
    _Float16* sW2 = (_Float16*)(smem + OFF_W2);
    _Float16* sA1 = (_Float16*)(smem + OFF_A1);
    _Float16* sA2 = (_Float16*)(smem + OFF_A2);
    float*    sG  = (float*)   (smem + OFF_G);
    float*    sB1 = (float*)   (smem + OFF_B1);
    float*    sB2 = (float*)   (smem + OFF_B2);
    float*    sWl = (float*)   (smem + OFF_WL);
    float*    sRd = (float*)   (smem + OFF_RED);

    const int tid  = threadIdx.x;
    const int lane = tid & 31;
    const int wave = tid >> 5;
    const int b0   = blockIdx.x * ROWS;

    // ---- one-shot LDS fill: 281.6 KB packed fp16 weights (L2-resident) -----
    {
        const uint4* src = (const uint4*)Wpack;
        uint4*       dst = (uint4*)sW1;                    // sW1||sW2 contiguous
        const int nvec = (NG_ * K1_ + NG_ * K2_) * 2 / 16; // 17600
        for (int i = tid; i < nvec; i += WGT) dst[i] = src[i];
        for (int i = tid; i < NG_; i += WGT) { sB1[i] = b1g[i]; sB2[i] = b2g[i]; }
        for (int i = tid; i <= H_; i += WGT) sWl[i] = wlg[i];
        // zero A buffers once: h=0 initial state and permanent K-padding
        for (int i = tid; i < ROWS * K1_; i += WGT) sA1[i] = (_Float16)0.0f;
        for (int i = tid; i < ROWS * K2_; i += WGT) sA2[i] = (_Float16)0.0f;
    }
    __syncthreads();

    // cell state in registers: 1600 (row,unit) pairs / 160 threads = 10 each
    float c1r[10], c2r[10];
#pragma unroll
    for (int i = 0; i < 10; ++i) { c1r[i] = 0.0f; c2r[i] = 0.0f; }

    // x double-buffer: threads 0..111 each own one (row, k<7) element
    const int xrow = tid / IN_;
    const int xk   = tid - xrow * IN_;
    const bool xldr = (tid < ROWS * IN_);
    float xreg = 0.0f;
    if (xldr) xreg = x[((size_t)(b0 + xrow) * T_ + 0) * IN_ + xk];

    for (int ts = 0; ts < T_; ++ts) {
        // ---- stage x_t into A1, then start x_{t+1} load ---------------------
        if (xldr) sA1[xrow * K1_ + xk] = (_Float16)xreg;
        __syncthreads();
        if (xldr && ts + 1 < T_)
            xreg = x[((size_t)(b0 + xrow) * T_ + (ts + 1)) * IN_ + xk];
        if (((ts & 7) == 0) && tid < ROWS && ts + 8 < T_)
            __builtin_prefetch(&x[((size_t)(b0 + tid) * T_ + (ts + 8)) * IN_], 0, 0);

        // ---- layer 1 gates: [x|h1] @ W1^T + b1 ------------------------------
        gemm_gates<K1_, K1_ / 32>(sA1, sW1, sB1, sG, lane, wave);
        __syncthreads();

        // ---- layer 1 elementwise (i,f,g,o -> c1,h1); bias already in gates --
#pragma unroll
        for (int ii = 0; ii < 10; ++ii) {
            int p = tid + ii * WGT;                    // 0..1599, no predicate
            int row = p / H_, u = p - row * H_;
            const float* g = sG + row * NG_;
            float iv = fast_sigmoid(g[u]);
            float fv = fast_sigmoid(g[H_ + u]);
            float gv = fast_tanh   (g[2 * H_ + u]);
            float ov = fast_sigmoid(g[3 * H_ + u]);
            float c  = fv * c1r[ii] + iv * gv;
            c1r[ii]  = c;
            _Float16 h = (_Float16)(ov * fast_tanh(c));
            sA1[row * K1_ + IN_ + u] = h;   // next step's layer-1 input
            sA2[row * K2_ + u]       = h;   // this step's layer-2 input
        }
        __syncthreads();

        // ---- layer 2 gates: [h1|h2] @ W2^T + b2 -----------------------------
        gemm_gates<K2_, K2_ / 32>(sA2, sW2, sB2, sG, lane, wave);
        __syncthreads();

        // ---- layer 2 elementwise (-> c2,h2) ---------------------------------
#pragma unroll
        for (int ii = 0; ii < 10; ++ii) {
            int p = tid + ii * WGT;
            int row = p / H_, u = p - row * H_;
            const float* g = sG + row * NG_;
            float iv = fast_sigmoid(g[u]);
            float fv = fast_sigmoid(g[H_ + u]);
            float gv = fast_tanh   (g[2 * H_ + u]);
            float ov = fast_sigmoid(g[3 * H_ + u]);
            float c  = fv * c2r[ii] + iv * gv;
            c2r[ii]  = c;
            sA2[row * K2_ + H_ + u] = (_Float16)(ov * fast_tanh(c));
        }
        __syncthreads();

        // ---- head: out[b,ts] = h2 . w_lin + b_lin (10 threads per row) ------
        {
            int row = tid / 10, seg = tid - row * 10;
            float p = 0.0f;
#pragma unroll
            for (int j = 0; j < 10; ++j)
                p += (float)sA2[row * K2_ + H_ + seg * 10 + j] * sWl[seg * 10 + j];
            sRd[tid] = p;
        }
        __syncthreads();
        if (tid < ROWS) {
            float s = sWl[H_];
#pragma unroll
            for (int j = 0; j < 10; ++j) s += sRd[tid * 10 + j];
            out[(size_t)(b0 + tid) * T_ + ts] = s;
        }
        // next write to sRd / sG is separated by later barriers; no sync here
    }
}

// ---------------------------------------------------------------------------
extern "C" void kernel_launch(void* const* d_in, const int* in_sizes, int n_in,
                              void* d_out, int out_size, void* d_ws, size_t ws_size,
                              hipStream_t stream) {
    const float* x    = (const float*)d_in[0];
    const float* Wih1 = (const float*)d_in[1];
    const float* Whh1 = (const float*)d_in[2];
    const float* bih1 = (const float*)d_in[3];
    const float* bhh1 = (const float*)d_in[4];
    const float* Wih2 = (const float*)d_in[5];
    const float* Whh2 = (const float*)d_in[6];
    const float* bih2 = (const float*)d_in[7];
    const float* bhh2 = (const float*)d_in[8];
    const float* Wlin = (const float*)d_in[9];
    const float* blin = (const float*)d_in[10];

    char* ws = (char*)d_ws;
    _Float16* W1c = (_Float16*)(ws);                 // 102400 B
    _Float16* W2c = (_Float16*)(ws + 102400);        // 179200 B
    float*    b1  = (float*)(ws + 281600);           // 1600 B
    float*    b2  = (float*)(ws + 283200);           // 1600 B
    float*    wl  = (float*)(ws + 284800);           // 101 floats (blin at [100])

    const int prep_total  = NG_ * K1_ + NG_ * K2_ + NG_ + NG_ + H_ + 1;
    const int prep_blocks = (prep_total + 255) / 256;
    prep_kernel<<<prep_blocks, 256, 0, stream>>>(Wih1, Whh1, bih1, bhh1,
                                                 Wih2, Whh2, bih2, bhh2,
                                                 Wlin, blin, W1c, W2c, b1, b2, wl);

    lstm_kernel<<<B_ / ROWS, WGT, SMEM_BYTES, stream>>>(
        x, W1c, b1, b2, wl, (float*)d_out);
}
